// MultiHeadAttention_33560874451577
// MI455X (gfx1250) — compile-verified, matrix operand order checked by reading, not run on hardware
//
#include <hip/hip_runtime.h>

// ---------------------------------------------------------------------------
// MHA for MI455X (gfx1250): bf16 WMMA GEMMs + flash attention (wave32).
// Softmax runs in the log2 domain: log2(e)/sqrt(depth) is folded into the
// Q projection so the hot loop uses raw v_exp_f32 (exp2) only.
// ---------------------------------------------------------------------------

#define H_   16
#define S_   2048
#define DM_  1024
#define DEP_ 64
#define B_   4
#define M_   (B_ * S_)   // 8192 tokens

typedef __attribute__((ext_vector_type(16))) __bf16 bf16x16;
typedef __attribute__((ext_vector_type(8)))  float  f32x8;
typedef unsigned short u16;
typedef unsigned int   u32;

static __device__ __forceinline__ u16 f2bf(float f) {
    __bf16 h = (__bf16)f;                      // hardware v_cvt
    return __builtin_bit_cast(u16, h);
}

// Build a 16x32-slice fragment from two 16-byte chunks (ISA bf16 A/B layout:
// per lane, halves 0..7 = K base, halves 8..15 = K base+16).
static __device__ __forceinline__ bf16x16 ld_frag(const u16* p0, const u16* p1) {
    union { bf16x16 f; uint4 q[2]; } u;
    u.q[0] = *reinterpret_cast<const uint4*>(p0);
    u.q[1] = *reinterpret_cast<const uint4*>(p1);
    return u.f;
}

#define WMMA_BF16(A, Bm, C) \
    __builtin_amdgcn_wmma_f32_16x16x32_bf16(false, (A), false, (Bm), (short)0, (C), false, false)

// ---------------------------------------------------------------------------
// fp32 -> bf16 (vectorized, grid-stride; element counts divisible by 4)
// ---------------------------------------------------------------------------
__global__ void cvt_f32_bf16(const float4* __restrict__ in, ushort2* __restrict__ out, long n4) {
    long i = (long)blockIdx.x * blockDim.x + threadIdx.x;
    long stride = (long)gridDim.x * blockDim.x;
    for (; i < n4; i += stride) {
        float4 v = in[i];
        out[2 * i]     = make_ushort2(f2bf(v.x), f2bf(v.y));
        out[2 * i + 1] = make_ushort2(f2bf(v.z), f2bf(v.w));
    }
}

// ---------------------------------------------------------------------------
// BF16 GEMM: Y[M,N] = (X[M,K] @ W[N,K]^T + bias[N]) * scale
// Block = 256 thr (8 waves) arranged 4(M) x 2(N): block tile 128 x 128,
// wave tile 32 x 64 (A fragments reused across 4 N-tiles, B across 2 M-tiles).
// MODE 0: bf16 per-head [b,h,s,64]            (Q with scale=log2e/8, K)
// MODE 1: bf16 per-head transposed [b,h,64,s] (V) — packed 16B stores
// MODE 2: fp32 row-major [M,N]                (final dense)
// ---------------------------------------------------------------------------
template <int MODE>
__global__ __launch_bounds__(256, 1)
void gemm_bf16(const u16* __restrict__ X, const u16* __restrict__ W,
               const float* __restrict__ bias, void* __restrict__ out,
               int K, float scale) {
    const int lane  = threadIdx.x & 31;
    const int wave  = threadIdx.x >> 5;
    const int l16   = lane & 15;
    const int hi    = lane >> 4;
    const int m0    = blockIdx.x * 128 + (wave >> 1) * 32;
    const int n0    = blockIdx.y * 128 + (wave & 1) * 64;

    f32x8 acc[2][4] = {};
    const u16* aRow0 = X + (size_t)(m0 + l16) * K;
    const u16* aRow1 = X + (size_t)(m0 + 16 + l16) * K;

    for (int kk = 0; kk < K; kk += 32) {
        bf16x16 af0 = ld_frag(aRow0 + kk + hi * 8, aRow0 + kk + 16 + hi * 8);
        bf16x16 af1 = ld_frag(aRow1 + kk + hi * 8, aRow1 + kk + 16 + hi * 8);
#pragma unroll
        for (int nt = 0; nt < 4; ++nt) {
            const u16* wRow = W + (size_t)(n0 + nt * 16 + l16) * K + kk;
            bf16x16 bfrag = ld_frag(wRow + hi * 8, wRow + 16 + hi * 8);
            acc[0][nt] = WMMA_BF16(af0, bfrag, acc[0][nt]);
            acc[1][nt] = WMMA_BF16(af1, bfrag, acc[1][nt]);
        }
    }

    // Epilogue. D layout: VGPR r at lane -> row mBase + r + 8*hi, col n0+nt*16+l16.
#pragma unroll
    for (int mt = 0; mt < 2; ++mt) {
        const int mb = m0 + mt * 16;
        if (MODE == 1) {
            u16* o = (u16*)out;
            const int b  = mb >> 11;
            const int sb = (mb & 2047) + hi * 8;
#pragma unroll
            for (int nt = 0; nt < 4; ++nt) {
                const int n = n0 + nt * 16 + l16;
                const int h = n >> 6, d = n & 63;
                const float bv = bias[n];
                union { u16 a[8]; uint4 q; } pk;
#pragma unroll
                for (int r = 0; r < 8; ++r) pk.a[r] = f2bf((acc[mt][nt][r] + bv) * scale);
                size_t idx = (((size_t)b * H_ + h) * DEP_ + d) * S_ + sb;
                *reinterpret_cast<uint4*>(o + idx) = pk.q;
            }
        } else {
#pragma unroll
            for (int nt = 0; nt < 4; ++nt) {
                const int n = n0 + nt * 16 + l16;
                const float bv = bias[n];
#pragma unroll
                for (int r = 0; r < 8; ++r) {
                    const int m = mb + r + hi * 8;
                    const float v = (acc[mt][nt][r] + bv) * scale;
                    if (MODE == 0) {
                        const int b = m >> 11, s = m & 2047;
                        const int h = n >> 6,  d = n & 63;
                        ((u16*)out)[(((size_t)b * H_ + h) * S_ + s) * DEP_ + d] = f2bf(v);
                    } else { // MODE 2
                        ((float*)out)[(size_t)m * DM_ + n] = v;
                    }
                }
            }
        }
    }
}

// ---------------------------------------------------------------------------
// Flash attention. One wave = 16 query rows x full depth(64).
// qH (pre-scaled by log2e/8 -> scores in log2 domain), kH: [bh][s][64] bf16;
// vT: [bh][64][s] bf16.  Row-sum of P via a 5th WMMA accumulator (all-ones B).
// Wave-uniform running max; alpha-rescale behind a uniform branch.
// All K and V fragments for the 32-key block are issued up front so the
// softmax VALU stretch hides their latency.
// grid = (S/128, B*H), block = 256 (8 waves).
// ---------------------------------------------------------------------------
__global__ __launch_bounds__(256, 1)
void flash_attn(const u16* __restrict__ qH, const u16* __restrict__ kH,
                const u16* __restrict__ vT, u16* __restrict__ ctx) {
    const int lane = threadIdx.x & 31;
    const int wave = threadIdx.x >> 5;
    const int l16  = lane & 15;
    const int hi   = lane >> 4;
    const int bh   = blockIdx.y;
    const int q0   = blockIdx.x * 128 + wave * 16;

    __shared__ __align__(16) u16 plds[8][16 * 32];   // per-wave P tile (row-major 16x32)
    u16* myP = plds[wave];

    const u16* qBase = qH + ((size_t)bh * S_ + q0 + l16) * DEP_;
    const bf16x16 qf0 = ld_frag(qBase + hi * 8,      qBase + 16 + hi * 8);   // d 0..31
    const bf16x16 qf1 = ld_frag(qBase + 32 + hi * 8, qBase + 48 + hi * 8);   // d 32..63

    bf16x16 ones;
    {
        union { bf16x16 f; u16 a[16]; } uo;
#pragma unroll
        for (int i = 0; i < 16; ++i) uo.a[i] = 0x3F80;   // bf16 1.0
        ones = uo.f;
    }

    f32x8 o[5] = {};        // o[0..3] = O tiles (depth), o[4] = row-sums of P
    float mrun = -3.0e38f;  // wave-uniform running max (log2 domain)

    const u16* kAll = kH + (size_t)bh * S_ * DEP_;
    const u16* vAll = vT + (size_t)bh * DEP_ * S_;

    for (int kb = 0; kb < S_; kb += 32) {
        // ---- issue ALL loads for this block up front (K then V) ----
        const u16* kr0 = kAll + (size_t)(kb + l16) * DEP_;
        const u16* kr1 = kAll + (size_t)(kb + 16 + l16) * DEP_;
        bf16x16 b00 = ld_frag(kr0 + hi * 8,      kr0 + 16 + hi * 8);
        bf16x16 b01 = ld_frag(kr0 + 32 + hi * 8, kr0 + 48 + hi * 8);
        bf16x16 b10 = ld_frag(kr1 + hi * 8,      kr1 + 16 + hi * 8);
        bf16x16 b11 = ld_frag(kr1 + 32 + hi * 8, kr1 + 48 + hi * 8);

        bf16x16 vf[4];
#pragma unroll
        for (int nt = 0; nt < 4; ++nt) {
            const u16* vr = vAll + (size_t)(nt * 16 + l16) * S_ + kb;
            vf[nt] = ld_frag(vr + hi * 8, vr + 16 + hi * 8);
        }

        // ---- scores (log2 domain; scale folded into Q) ----
        f32x8 s0 = {}, s1 = {};
        s0 = WMMA_BF16(qf0, b00, s0);
        s0 = WMMA_BF16(qf1, b01, s0);
        s1 = WMMA_BF16(qf0, b10, s1);
        s1 = WMMA_BF16(qf1, b11, s1);

        // ---- wave-uniform tile max ----
        float mx = fmaxf(s0[0], s1[0]);
#pragma unroll
        for (int r = 1; r < 8; ++r) mx = fmaxf(mx, fmaxf(s0[r], s1[r]));
        mx = fmaxf(mx, __shfl_xor(mx, 1, 32));
        mx = fmaxf(mx, __shfl_xor(mx, 2, 32));
        mx = fmaxf(mx, __shfl_xor(mx, 4, 32));
        mx = fmaxf(mx, __shfl_xor(mx, 8, 32));
        mx = fmaxf(mx, __shfl_xor(mx, 16, 32));

        if (mx > mrun) {                       // uniform branch: rare after warm-up
            const float alpha = __builtin_amdgcn_exp2f(mrun - mx);
            mrun = mx;
#pragma unroll
            for (int t = 0; t < 5; ++t)
#pragma unroll
                for (int r = 0; r < 8; ++r) o[t][r] *= alpha;
        }

        // ---- P = exp2(S - m), staged to LDS in A-fragment order ----
#pragma unroll
        for (int r = 0; r < 8; ++r) {
            const int row = r + hi * 8;
            myP[row * 32 + l16]      = f2bf(__builtin_amdgcn_exp2f(s0[r] - mrun));
            myP[row * 32 + 16 + l16] = f2bf(__builtin_amdgcn_exp2f(s1[r] - mrun));
        }

        // ---- O += P @ V ;  o[4] += P @ ones (row sums) ----
        bf16x16 pf = ld_frag(myP + l16 * 32 + hi * 8, myP + l16 * 32 + 16 + hi * 8);
#pragma unroll
        for (int nt = 0; nt < 4; ++nt) o[nt] = WMMA_BF16(pf, vf[nt], o[nt]);
        o[4] = WMMA_BF16(pf, ones, o[4]);
    }

    // ---- epilogue: normalize by row sums, store context [token][1024] ----
    const int b = bh >> 4, h = bh & 15;
#pragma unroll
    for (int r = 0; r < 8; ++r) {
        const float inv = __builtin_amdgcn_rcpf(o[4][r]);
        const int tok = b * S_ + q0 + r + hi * 8;
#pragma unroll
        for (int nt = 0; nt < 4; ++nt) {
            const int col = h * DEP_ + nt * 16 + l16;
            ctx[(size_t)tok * DM_ + col] = f2bf(o[nt][r] * inv);
        }
    }
}

// ---------------------------------------------------------------------------
// launch
// ---------------------------------------------------------------------------
extern "C" void kernel_launch(void* const* d_in, const int* in_sizes, int n_in,
                              void* d_out, int out_size, void* d_ws, size_t ws_size,
                              hipStream_t stream) {
    const float* query   = (const float*)d_in[0];
    const float* key     = (const float*)d_in[1];
    const float* value   = (const float*)d_in[2];
    const float* wq_w    = (const float*)d_in[3];
    const float* wq_b    = (const float*)d_in[4];
    const float* wk_w    = (const float*)d_in[5];
    const float* wk_b    = (const float*)d_in[6];
    const float* wv_w    = (const float*)d_in[7];
    const float* wv_b    = (const float*)d_in[8];
    const float* dense_w = (const float*)d_in[9];
    const float* dense_b = (const float*)d_in[10];

    const size_t MK = (size_t)M_ * DM_;     // 8M activation elems
    const size_t WK = (size_t)DM_ * DM_;    // 1M weight elems

    u16* w = (u16*)d_ws;
    u16* xq = w;                 // bf16 inputs
    u16* xk = xq + MK;
    u16* xv = xk + MK;
    u16* bq = xv + MK;           // bf16 weights
    u16* bk = bq + WK;
    u16* bv = bk + WK;
    u16* bd = bv + WK;
    u16* qH = bd + WK;           // [b,h,s,64]  (pre-scaled by log2e/8)
    u16* kH = qH + MK;
    u16* vT = kH + MK;           // [b,h,64,s]
    u16* cx = vT + MK;           // context [token][1024]

    // 1) fp32 -> bf16
    cvt_f32_bf16<<<1024, 256, 0, stream>>>((const float4*)query,   (ushort2*)xq, (long)(MK / 4));
    cvt_f32_bf16<<<1024, 256, 0, stream>>>((const float4*)key,     (ushort2*)xk, (long)(MK / 4));
    cvt_f32_bf16<<<1024, 256, 0, stream>>>((const float4*)value,   (ushort2*)xv, (long)(MK / 4));
    cvt_f32_bf16<<<256,  256, 0, stream>>>((const float4*)wq_w,    (ushort2*)bq, (long)(WK / 4));
    cvt_f32_bf16<<<256,  256, 0, stream>>>((const float4*)wk_w,    (ushort2*)bk, (long)(WK / 4));
    cvt_f32_bf16<<<256,  256, 0, stream>>>((const float4*)wv_w,    (ushort2*)bv, (long)(WK / 4));
    cvt_f32_bf16<<<256,  256, 0, stream>>>((const float4*)dense_w, (ushort2*)bd, (long)(WK / 4));

    // 2) projections (Q pre-scaled by log2(e)/sqrt(depth) -> log2-domain scores)
    const float QSCALE = 0.125f * 1.44269504088896340736f;
    dim3 gg(M_ / 128, DM_ / 128);
    gemm_bf16<0><<<gg, 256, 0, stream>>>(xq, bq, wq_b, qH, DM_, QSCALE);
    gemm_bf16<0><<<gg, 256, 0, stream>>>(xk, bk, wk_b, kH, DM_, 1.0f);
    gemm_bf16<1><<<gg, 256, 0, stream>>>(xv, bv, wv_b, vT, DM_, 1.0f);

    // 3) flash attention
    flash_attn<<<dim3(S_ / 128, B_ * H_), 256, 0, stream>>>(qH, kH, vT, cx);

    // 4) output projection (fp32 out)
    gemm_bf16<2><<<gg, 256, 0, stream>>>(cx, bd, dense_b, d_out, DM_, 1.0f);
}
